// DiffpoolMessageBlock_1683627180253
// MI455X (gfx1250) — compile-verified, matrix operand order checked by reading, not run on hardware
//
#include <hip/hip_runtime.h>
#include <hip/hip_bf16.h>
#include <math.h>

typedef __attribute__((ext_vector_type(16))) __bf16 v16bf;
typedef __attribute__((ext_vector_type(8)))  float  v8f;

#define PI_F   3.14159265358979323846f
#define W_CUT  0.62831853071795864769f   // pi / CUTOFF, CUTOFF = 5

// K index held at fragment slot i for a lane in half `hi` (16-bit A/B 16x16x32 layout):
// lanes 0-15 hold K {0..7, 16..23}, lanes 16-31 hold K {8..15, 24..31}
__device__ __forceinline__ int kslot(int i, int hi) {
    return (i < 8) ? (hi * 8 + i) : (16 + hi * 8 + (i - 8));
}

// ---------------------------------------------------------------- zero output
__global__ void zero_f32(float* __restrict__ p, long n) {
    long i = (long)blockIdx.x * blockDim.x + threadIdx.x;
    if (i < n) p[i] = 0.0f;
}

// ------------------------------------------- pack Wd (20x512 f32) into bf16 B fragments
// 32 column tiles of 16; entry (tile t, lane l) = 16 bf16 values, K zero-padded to 32.
__global__ void pack_wd(const float* __restrict__ Wd, __bf16* __restrict__ out) {
    for (int ent = threadIdx.x; ent < 1024; ent += blockDim.x) {
        int t  = ent >> 5;
        int l  = ent & 31;
        int hi = l >> 4;
        int n  = t * 16 + (l & 15);
#pragma unroll
        for (int i = 0; i < 16; ++i) {
            int k = kslot(i, hi);
            float v = (k < 20) ? Wd[k * 512 + n] : 0.0f;
            out[ent * 16 + i] = (__bf16)v;
        }
    }
}

// ------------------------------------------- pack generic B (K x NC, K%32==0) into fragments
// entry index = ((kt * (NC/16) + nt) * 32 + lane), 16 bf16 per entry.
__global__ void pack_b(const float* __restrict__ B, __bf16* __restrict__ out, int K, int NC) {
    int ntiles = NC >> 4;
    int total  = (K >> 5) * ntiles * 32;
    for (int ent = blockIdx.x * blockDim.x + threadIdx.x; ent < total;
         ent += gridDim.x * blockDim.x) {
        int l  = ent & 31;
        int nt = (ent >> 5) % ntiles;
        int kt = (ent >> 5) / ntiles;
        int hi = l >> 4;
        int n  = nt * 16 + (l & 15);
#pragma unroll
        for (int i = 0; i < 16; ++i) {
            int k = kt * 32 + kslot(i, hi);
            out[(size_t)ent * 16 + i] = (__bf16)B[(size_t)k * NC + n];
        }
    }
}

// ---------------------------------------------------------------- bf16 WMMA GEMM
// C(M x NC) = act(A(M x K) @ B + bias), B pre-packed into bf16 fragments. act=1 -> silu.
// Block: 256 thr = 8 waves; each wave one 16x16 tile; block covers 128 rows x 16 cols.
__global__ void gemm_bf16_wmma(const float* __restrict__ A, const v16bf* __restrict__ Bp,
                               const float* __restrict__ bias, float* __restrict__ C,
                               int M, int K, int NC, int act) {
    int lane = threadIdx.x & 31;
    int wave = threadIdx.x >> 5;
    int m1   = lane & 15;
    int hi   = lane >> 4;
    int row0 = blockIdx.x * 128 + wave * 16;
    int col0 = blockIdx.y * 16;
    int arow = row0 + m1; if (arow >= M) arow = M - 1;   // clamp loads, guard stores
    int n      = col0 + m1;
    int ntiles = NC >> 4;

    v8f acc = {};
    for (int kb = 0; kb < K; kb += 32) {
        int kt  = kb >> 5;
        int ka0 = kb + hi * 8;
        int ka1 = kb + 16 + hi * 8;
        const float4* ap0 = reinterpret_cast<const float4*>(A + (size_t)arow * K + ka0);
        const float4* ap1 = reinterpret_cast<const float4*>(A + (size_t)arow * K + ka1);
        float4 a0 = ap0[0], a1 = ap0[1], a2 = ap1[0], a3 = ap1[1];
        v16bf af;
        af[0]=(__bf16)a0.x; af[1]=(__bf16)a0.y; af[2]=(__bf16)a0.z; af[3]=(__bf16)a0.w;
        af[4]=(__bf16)a1.x; af[5]=(__bf16)a1.y; af[6]=(__bf16)a1.z; af[7]=(__bf16)a1.w;
        af[8]=(__bf16)a2.x; af[9]=(__bf16)a2.y; af[10]=(__bf16)a2.z; af[11]=(__bf16)a2.w;
        af[12]=(__bf16)a3.x; af[13]=(__bf16)a3.y; af[14]=(__bf16)a3.z; af[15]=(__bf16)a3.w;

        v16bf bf = Bp[(size_t)(kt * ntiles + blockIdx.y) * 32 + lane];

        acc = __builtin_amdgcn_wmma_f32_16x16x32_bf16(false, af, false, bf,
                                                      (short)0, acc, false, false);
    }

    float bv = bias[n];
#pragma unroll
    for (int r = 0; r < 8; ++r) {
        int m = row0 + hi * 8 + r;
        if (m < M) {
            float v = acc[r] + bv;
            if (act) v = __fdividef(v, 1.0f + __expf(-v));   // fast silu
            C[(size_t)m * NC + n] = v;
        }
    }
}

// ---------------------------------------------------------------- fused edge kernel
// Per wave: 16 edges. RBF(16x32 bf16, K padded) @ Wd tiles (LDS) via WMMA -> w_s,
// then gather phi/v_j, cross product, atomic scatter into ds_i / dv_i.
__global__ void edge_wmma(const float* __restrict__ r_ij, const int* __restrict__ nbrs,
                          const float* __restrict__ cg,   const float* __restrict__ phi,
                          const float* __restrict__ v_j,  const uint4* __restrict__ wdpack,
                          const float* __restrict__ bd,   float* __restrict__ ds_i,
                          float* __restrict__ dv_i, int E, int N) {
    __shared__ __align__(32) unsigned char ldsb[32768];
    uint4* l4 = (uint4*)ldsb;
    for (int i = threadIdx.x; i < 2048; i += blockDim.x) l4[i] = wdpack[i];
    __syncthreads();
    const v16bf* Bt = (const v16bf*)ldsb;

    int lane = threadIdx.x & 31;
    int wave = threadIdx.x >> 5;
    int m1   = lane & 15;
    int hi   = lane >> 4;
    long e0  = ((long)blockIdx.x * 8 + wave) * 16;
    if (e0 >= E) return;

    long e = e0 + m1;
    bool valid = (e < E);
    if (!valid) e = E - 1;

    float rx = r_ij[e * 3 + 0], ry = r_ij[e * 3 + 1], rz = r_ij[e * 3 + 2];
    float s2   = rx * rx + ry * ry + rz * rz + 1e-15f;
    float invd = __frsqrt_rn(s2);          // 1/d
    float d    = s2 * invd;                // d
    float ux = rx * invd, uy = ry * invd, uz = rz * invd;
    float env = (d < 5.0f) ? 0.5f * (__cosf(PI_F * d * 0.2f) + 1.0f) : 0.0f;
    int src = nbrs[2 * e + 0];
    int dst = nbrs[2 * e + 1];
    float wgt = valid ? cg[(size_t)src * N + dst] : 0.0f;

    // A fragment: rbf_k = sin((k+1)*pi/5*d)/d for k<20, zero-padded to K=32
    v16bf af;
#pragma unroll
    for (int i = 0; i < 16; ++i) {
        int k = kslot(i, hi);
        float v = (k < 20) ? __sinf((float)(k + 1) * W_CUT * d) * invd : 0.0f;
        af[i] = (__bf16)v;
    }

    // broadcast per-edge scalars for the 8 C rows this lane covers
    float envR[8], wR[8], uxR[8], uyR[8], uzR[8];
    int srcR[8], dstR[8];
#pragma unroll
    for (int r = 0; r < 8; ++r) {
        int m = hi * 8 + r;
        envR[r] = __shfl(env, m, 32);
        wR[r]   = __shfl(wgt, m, 32);
        uxR[r]  = __shfl(ux,  m, 32);
        uyR[r]  = __shfl(uy,  m, 32);
        uzR[r]  = __shfl(uz,  m, 32);
        srcR[r] = __shfl(src, m, 32);
        dstR[r] = __shfl(dst, m, 32);
    }

    v8f zero = {};
    for (int ft = 0; ft < 8; ++ft) {       // splits 0..3 live at column tiles ft,8+ft,16+ft,24+ft
        v16bf b0 = Bt[(ft)      * 32 + lane];
        v16bf b1 = Bt[(8  + ft) * 32 + lane];
        v16bf b2 = Bt[(16 + ft) * 32 + lane];
        v16bf b3 = Bt[(24 + ft) * 32 + lane];
        v8f c0 = __builtin_amdgcn_wmma_f32_16x16x32_bf16(false, af, false, b0, (short)0, zero, false, false);
        v8f c1 = __builtin_amdgcn_wmma_f32_16x16x32_bf16(false, af, false, b1, (short)0, zero, false, false);
        v8f c2 = __builtin_amdgcn_wmma_f32_16x16x32_bf16(false, af, false, b2, (short)0, zero, false, false);
        v8f c3 = __builtin_amdgcn_wmma_f32_16x16x32_bf16(false, af, false, b3, (short)0, zero, false, false);

        int f = ft * 16 + m1;
        float bd0 = bd[f], bd1 = bd[128 + f], bd2 = bd[256 + f], bd3 = bd[384 + f];
#pragma unroll
        for (int r = 0; r < 8; ++r) {
            float envm = envR[r], wm = wR[r];
            int sm = srcR[r], dm = dstR[r];
            const float* ph = phi + (size_t)dm * 512;
            float i0 = (c0[r] + bd0) * envm * ph[f];
            float i1 = (c1[r] + bd1) * envm * ph[128 + f];
            float i2 = (c2[r] + bd2) * envm * ph[256 + f];
            float i3 = (c3[r] + bd3) * envm * ph[384 + f];
            const float* vd = v_j + ((size_t)dm * 128 + f) * 3;
            const float* vs = v_j + ((size_t)sm * 128 + f) * 3;
            float dx = vd[0], dy = vd[1], dz = vd[2];
            float sx = vs[0], sy = vs[1], sz = vs[2];
            float cx = sy * dz - sz * dy;
            float cy = sz * dx - sx * dz;
            float cz = sx * dy - sy * dx;
            float gx = i2 * uxR[r] + i0 * dx + i3 * cx;
            float gy = i2 * uyR[r] + i0 * dy + i3 * cy;
            float gz = i2 * uzR[r] + i0 * dz + i3 * cz;
            float* dvp = dv_i + ((size_t)sm * 128 + f) * 3;
            atomicAdd(dvp + 0, gx * wm);
            atomicAdd(dvp + 1, gy * wm);
            atomicAdd(dvp + 2, gz * wm);
            atomicAdd(ds_i + (size_t)sm * 128 + f, i1 * wm);
        }
    }
}

// ---------------------------------------------------------------- launcher
extern "C" void kernel_launch(void* const* d_in, const int* in_sizes, int n_in,
                              void* d_out, int out_size, void* d_ws, size_t ws_size,
                              hipStream_t stream) {
    (void)n_in; (void)ws_size;
    const float* s_j  = (const float*)d_in[0];
    const float* v_j  = (const float*)d_in[1];
    const float* r_ij = (const float*)d_in[2];
    const int*   nbrs = (const int*)d_in[3];
    const float* cg   = (const float*)d_in[4];
    const float* W1   = (const float*)d_in[5];
    const float* b1   = (const float*)d_in[6];
    const float* W2   = (const float*)d_in[7];
    const float* b2   = (const float*)d_in[8];
    const float* Wd   = (const float*)d_in[9];
    const float* bd   = (const float*)d_in[10];
    float* out = (float*)d_out;

    const int F = 128;
    const int N = in_sizes[0] / F;      // 10000
    const int E = in_sizes[2] / 3;      // 320000

    // workspace layout (bytes):
    //   [0,      32K)  wdpack  : 32 tiles * 32 lanes * 16 bf16
    //   [32K,    64K)  w1pack  : (128/32)*(128/16)*32*16 bf16 = 32 KB
    //   [64K,   192K)  w2pack  : (128/32)*(512/16)*32*16 bf16 = 128 KB
    //   [192K,  ...)   h       : N*F  f32
    //   [...,   ...)   phi     : N*4F f32
    char*   ws     = (char*)d_ws;
    __bf16* wdpack = (__bf16*)(ws);
    __bf16* w1pack = (__bf16*)(ws + (32 << 10));
    __bf16* w2pack = (__bf16*)(ws + (64 << 10));
    float*  h      = (float*)(ws + (192 << 10));
    float*  phi    = (float*)(ws + (192 << 10) + (size_t)N * F * sizeof(float));

    long outn = (long)out_size;
    zero_f32<<<dim3((unsigned)((outn + 255) / 256)), 256, 0, stream>>>(out, outn);
    pack_wd<<<1, 256, 0, stream>>>(Wd, wdpack);
    pack_b <<<4, 256, 0, stream>>>(W1, w1pack, F, F);       // 1024 entries
    pack_b <<<16, 256, 0, stream>>>(W2, w2pack, F, 4 * F);  // 4096 entries

    dim3 g1((N + 127) / 128, F / 16);
    gemm_bf16_wmma<<<g1, 256, 0, stream>>>(s_j, (const v16bf*)w1pack, b1, h, N, F, F, 1);
    dim3 g2((N + 127) / 128, (4 * F) / 16);
    gemm_bf16_wmma<<<g2, 256, 0, stream>>>(h, (const v16bf*)w2pack, b2, phi, N, F, 4 * F, 0);

    int nblk = (E + 127) / 128;
    edge_wmma<<<nblk, 256, 0, stream>>>(r_ij, nbrs, cg, phi, v_j, (const uint4*)wdpack,
                                        bd, out, out + (size_t)N * F, E, N);
}
